// DeFUM_53755810677190
// MI455X (gfx1250) — compile-verified, matrix operand order checked by reading, not run on hardware
//
#include <hip/hip_runtime.h>

typedef __attribute__((ext_vector_type(16))) __bf16 v16bf;
typedef __attribute__((ext_vector_type(8)))  __bf16 v8bf;
typedef __attribute__((ext_vector_type(8)))  float  v8f;

namespace {

constexpr int kB   = 64;
constexpr int kN   = 256;     // 64 ocr + 192 obj tokens
constexpr int kD   = 768;
constexpr int kH   = 12;
constexpr int kDH  = 64;
constexpr int kL   = 4;
constexpr int kDFF = 2048;
constexpr long long kT   = (long long)kB * kN;   // 16384 tokens
constexpr long long kTD  = kT * kD;              // 12,582,912
constexpr long long kTD3 = kT * 3 * kD;

// ---- workspace layout (bytes) ----
constexpr size_t OFF_XF  = 0;                          // residual stream f32, T*D
constexpr size_t OFF_XB  = OFF_XF + (size_t)kTD * 4;   // bf16 copy of stream
constexpr size_t OFF_DV  = OFF_XB + (size_t)kTD * 2;   // log-depth, T f32
constexpr size_t OFF_WB  = OFF_DV + (size_t)kT * 4;    // all weights bf16
// weight sub-offsets in bf16 elements
constexpr long long WB_SAWQ = 0;
constexpr long long WB_SAWK = 589824;
constexpr long long WB_SAWV = 1179648;
constexpr long long WB_QKV  = 1769472;   // 4 * 2304*768
constexpr long long WB_OUT  = 8847360;   // 4 * 768*768
constexpr long long WB_FF1  = 11206656;  // 4 * 2048*768
constexpr long long WB_FF2  = 17498112;  // 4 * 768*2048
constexpr long long WB_TOT  = 23789568;
constexpr size_t OFF_QKV = OFF_WB  + (size_t)WB_TOT * 2;   // bf16 qkv acts, T*3D
constexpr size_t OFF_AO  = OFF_QKV + (size_t)kTD3 * 2;     // bf16 attn out, T*D
constexpr size_t OFF_Y   = OFF_AO  + (size_t)kTD * 2;      // f32 GEMM out, T*D
constexpr size_t OFF_S   = OFF_Y   + (size_t)kTD * 4;      // f32 scores (201MB)
constexpr size_t OFF_P   = OFF_S   + (size_t)kB * kH * kN * kN * 4; // bf16 probs / gelu out

// ---------------- WMMA GEMM ----------------
// C[M,Nc] (+bias, *scale, opt GELU) = A[M,K] * op(B)
//   op(B)=B^T (B stored [Nc,K]) if !TRANSB ; op(B)=B (B stored [K,Nc]) if TRANSB
// Batched over blockIdx.z with two-level strides: off = (z/zdiv)*z1 + (z%zdiv)*z2.
// Block = 256 threads = 8 waves; each wave owns a 32x64 output strip
// (2 M-frags x 4 N-frags = 8 WMMA per K-step; B fragments reused across M).
constexpr int BM = 256, BN = 64, BK = 32, LDSK = 40; // 40 bf16 = 80B padded row

template <bool TRANSB, bool OUTBF, bool GELU>
__global__ __launch_bounds__(256) void gemm_wmma(
    const __bf16* __restrict__ A, int lda, long long zA1, long long zA2,
    const __bf16* __restrict__ Bm, int ldb, long long zB1, long long zB2,
    void* __restrict__ Cp, int ldc, long long zC1, long long zC2,
    const float* __restrict__ bias, int K, float scale, int zdiv) {
  __shared__ __align__(16) __bf16 As[BM * LDSK];
  __shared__ __align__(16) __bf16 Bs[BN * LDSK];

  const int z  = blockIdx.z;
  const int zq = z / zdiv, zr = z % zdiv;
  const long long az = (long long)zq * zA1 + (long long)zr * zA2;
  const long long bz = (long long)zq * zB1 + (long long)zr * zB2;
  const long long cz = (long long)zq * zC1 + (long long)zr * zC2;

  const __bf16* Ab = A + az + (long long)blockIdx.y * BM * lda;
  const __bf16* Bb = Bm + bz;
  const int colBlock = blockIdx.x * BN;

  const int tid  = threadIdx.x;
  const int lane = tid & 31;
  const int wv   = tid >> 5;       // 8 waves
  const int lrow = lane & 15;
  const int hi   = (lane >> 4) & 1;
  const int wrow = wv * 32;        // wave's 32 output rows

  v8f acc[2][4];
#pragma unroll
  for (int m = 0; m < 2; ++m)
#pragma unroll
    for (int j = 0; j < 4; ++j)
#pragma unroll
      for (int r = 0; r < 8; ++r) acc[m][j][r] = 0.0f;

  for (int k0 = 0; k0 < K; k0 += BK) {
    // A tile: 256x32 bf16, 8 elements per thread x4 iterations, vectorized
#pragma unroll
    for (int it = 0; it < 4; ++it) {
      int e = (tid + it * 256) * 8;
      int r = e >> 5, c = e & 31;
      *(v8bf*)&As[r * LDSK + c] =
          *(const v8bf*)&Ab[(long long)r * lda + k0 + c];
    }
    // B tile -> Bs[n][k] (64x32)
    if (!TRANSB) {
      int e = tid * 8;
      int r = e >> 5, c = e & 31;
      *(v8bf*)&Bs[r * LDSK + c] =
          *(const v8bf*)&Bb[(long long)(colBlock + r) * ldb + k0 + c];
    } else {
#pragma unroll
      for (int it = 0; it < 8; ++it) {
        int e = tid + it * 256;
        int kk = e >> 6, n = e & 63;   // consecutive lanes -> consecutive n (coalesced)
        Bs[n * LDSK + kk] = Bb[(long long)(k0 + kk) * ldb + colBlock + n];
      }
    }
    __syncthreads();

    // A fragments: 16-bit A 16x32 layout (lanes 0-15: K0-7/16-23, lanes 16-31: K8-15/24-31)
    v16bf af[2];
#pragma unroll
    for (int m = 0; m < 2; ++m) {
      const v8bf alo = *(const v8bf*)&As[(wrow + m * 16 + lrow) * LDSK + (hi ? 8 : 0)];
      const v8bf ahi = *(const v8bf*)&As[(wrow + m * 16 + lrow) * LDSK + (hi ? 24 : 16)];
#pragma unroll
      for (int e = 0; e < 8; ++e) { af[m][e] = alo[e]; af[m][e + 8] = ahi[e]; }
    }
#pragma unroll
    for (int j = 0; j < 4; ++j) {
      v16bf bfv;
      const v8bf blo = *(const v8bf*)&Bs[(j * 16 + lrow) * LDSK + (hi ? 8 : 0)];
      const v8bf bhi = *(const v8bf*)&Bs[(j * 16 + lrow) * LDSK + (hi ? 24 : 16)];
#pragma unroll
      for (int e = 0; e < 8; ++e) { bfv[e] = blo[e]; bfv[e + 8] = bhi[e]; }
#pragma unroll
      for (int m = 0; m < 2; ++m)
        acc[m][j] = __builtin_amdgcn_wmma_f32_16x16x32_bf16(
            false, af[m], false, bfv, (short)0, acc[m][j], false, false);
    }
    __syncthreads();
  }

  // C/D layout: lanes 0-15 -> M = r, lanes 16-31 -> M = 8+r; N = lane%16
#pragma unroll
  for (int m = 0; m < 2; ++m) {
    const long long crow0 =
        (long long)blockIdx.y * BM + wrow + m * 16 + (hi ? 8 : 0);
#pragma unroll
    for (int j = 0; j < 4; ++j) {
      const int col = colBlock + j * 16 + lrow;
      const float bv = bias ? bias[col] : 0.0f;
#pragma unroll
      for (int r = 0; r < 8; ++r) {
        const long long idx = cz + (crow0 + r) * (long long)ldc + col;
        float v = acc[m][j][r] * scale + bv;
        if (GELU) v = 0.5f * v * (1.0f + erff(v * 0.70710678118654752f));
        if (OUTBF) ((__bf16*)Cp)[idx] = (__bf16)v;
        else       ((float*)Cp)[idx]  = v;
      }
    }
  }
}

// ---------------- elementwise / reduction kernels ----------------
__device__ inline float wred_sum(float v) {
#pragma unroll
  for (int o = 16; o > 0; o >>= 1) v += __shfl_xor(v, o);
  return v;
}
__device__ inline float wred_max(float v) {
#pragma unroll
  for (int o = 16; o > 0; o >>= 1) v = fmaxf(v, __shfl_xor(v, o));
  return v;
}

__global__ void concat_x(const float* __restrict__ ocr, const float* __restrict__ obj,
                         float* __restrict__ xf, __bf16* __restrict__ xb) {
  long long i = (long long)blockIdx.x * 256 + threadIdx.x;
  if (i >= kTD) return;
  int d = (int)(i % kD);
  long long t = i / kD;
  int b = (int)(t / kN), n = (int)(t % kN);
  float v = (n < 64) ? ocr[((long long)b * 64 + n) * kD + d]
                     : obj[((long long)b * 192 + (n - 64)) * kD + d];
  xf[i] = v;
  xb[i] = (__bf16)v;
}

__global__ void dvlog_k(const float* __restrict__ odv, const float* __restrict__ bdv,
                        float* __restrict__ ldv) {
  int i = blockIdx.x * 256 + threadIdx.x;
  if (i >= (int)kT) return;
  int b = i / kN, n = i % kN;
  float v = (n < 64) ? odv[b * 64 + n] : bdv[b * 192 + (n - 64)];
  ldv[i] = logf(v);
}

__global__ void cvt_f32_bf16(const float* __restrict__ s, __bf16* __restrict__ d, long long n) {
  long long i = (long long)blockIdx.x * 256 + threadIdx.x;
  if (i < n) d[i] = (__bf16)s[i];
}

// softmax over 256-wide rows; one wave per row; optional relative-depth bias
template <bool SEM>
__global__ __launch_bounds__(256) void softmax_k(const float* __restrict__ S,
                                                 __bf16* __restrict__ P,
                                                 const float* __restrict__ ldv) {
  const int wv = threadIdx.x >> 5, lane = threadIdx.x & 31;
  const long long row = (long long)blockIdx.x * 8 + wv;
  const float* sp = S + row * kN;
  float li = 0.0f;
  int bT = 0;
  if (SEM) { bT = (int)(row >> 8) << 8; li = ldv[row]; }
  float vals[8];
  float m = -3.0e30f;
#pragma unroll
  for (int j = 0; j < 8; ++j) {
    int c = lane + j * 32;
    float v = sp[c];
    if (SEM) v += fabsf(li - ldv[bT + c]);
    vals[j] = v;
    m = fmaxf(m, v);
  }
  m = wred_max(m);
  float s = 0.0f;
#pragma unroll
  for (int j = 0; j < 8; ++j) { vals[j] = expf(vals[j] - m); s += vals[j]; }
  s = wred_sum(s);
  const float inv = 1.0f / s;
#pragma unroll
  for (int j = 0; j < 8; ++j) P[row * kN + lane + j * 32] = (__bf16)(vals[j] * inv);
}

// h = LN(res + y), writes f32 + bf16 copies. One block per token row (768 = 3*256).
__global__ __launch_bounds__(256) void ln_resid(const float* __restrict__ res,
                                                const float* __restrict__ y,
                                                const float* __restrict__ g,
                                                const float* __restrict__ bta,
                                                float* __restrict__ of,
                                                __bf16* __restrict__ ob) {
  __shared__ float red[8];
  const long long row = blockIdx.x;
  const int t = threadIdx.x, lane = t & 31, wv = t >> 5;
  const float* rp = res + row * kD;
  const float* yp = y + row * kD;
  float v[3];
  float s = 0.0f;
#pragma unroll
  for (int i = 0; i < 3; ++i) { int c = t + i * 256; v[i] = rp[c] + yp[c]; s += v[i]; }
  s = wred_sum(s);
  if (lane == 0) red[wv] = s;
  __syncthreads();
  float tot = 0.0f;
#pragma unroll
  for (int i = 0; i < 8; ++i) tot += red[i];
  const float mean = tot * (1.0f / kD);
  __syncthreads();
  float q = 0.0f;
#pragma unroll
  for (int i = 0; i < 3; ++i) { float d = v[i] - mean; q += d * d; }
  q = wred_sum(q);
  if (lane == 0) red[wv] = q;
  __syncthreads();
  float qt = 0.0f;
#pragma unroll
  for (int i = 0; i < 8; ++i) qt += red[i];
  const float rstd = rsqrtf(qt * (1.0f / kD) + 1e-5f);
#pragma unroll
  for (int i = 0; i < 3; ++i) {
    int c = t + i * 256;
    float o = (v[i] - mean) * rstd * g[c] + bta[c];
    of[row * kD + c] = o;
    ob[row * kD + c] = (__bf16)o;
  }
}

__global__ void slice_k(const float* __restrict__ xf, float* __restrict__ out) {
  long long i = (long long)blockIdx.x * 256 + threadIdx.x;
  if (i >= (long long)kB * 64 * kD) return;
  int d = (int)(i % kD);
  long long t = i / kD;
  int b = (int)(t / 64), n = (int)(t % 64);
  out[i] = xf[((long long)b * kN + n) * kD + d];
}

// ---------------- host dispatch helper ----------------
static void gemm(hipStream_t st, bool transB, bool outbf, bool gelu,
                 const __bf16* A, int lda, long long zA1, long long zA2,
                 const __bf16* Bm, int ldb, long long zB1, long long zB2,
                 void* C, int ldc, long long zC1, long long zC2,
                 const float* bias, int M, int Nc, int K, float scale, int zdiv, int nz) {
  dim3 grd(Nc / BN, M / BM, nz), blk(256);
  if (gelu)
    gemm_wmma<false, true, true><<<grd, blk, 0, st>>>(A, lda, zA1, zA2, Bm, ldb, zB1, zB2, C, ldc, zC1, zC2, bias, K, scale, zdiv);
  else if (!transB && !outbf)
    gemm_wmma<false, false, false><<<grd, blk, 0, st>>>(A, lda, zA1, zA2, Bm, ldb, zB1, zB2, C, ldc, zC1, zC2, bias, K, scale, zdiv);
  else if (!transB && outbf)
    gemm_wmma<false, true, false><<<grd, blk, 0, st>>>(A, lda, zA1, zA2, Bm, ldb, zB1, zB2, C, ldc, zC1, zC2, bias, K, scale, zdiv);
  else if (transB && !outbf)
    gemm_wmma<true, false, false><<<grd, blk, 0, st>>>(A, lda, zA1, zA2, Bm, ldb, zB1, zB2, C, ldc, zC1, zC2, bias, K, scale, zdiv);
  else
    gemm_wmma<true, true, false><<<grd, blk, 0, st>>>(A, lda, zA1, zA2, Bm, ldb, zB1, zB2, C, ldc, zC1, zC2, bias, K, scale, zdiv);
}

}  // namespace

extern "C" void kernel_launch(void* const* d_in, const int* in_sizes, int n_in,
                              void* d_out, int out_size, void* d_ws, size_t ws_size,
                              hipStream_t stream) {
  (void)in_sizes; (void)n_in; (void)out_size; (void)ws_size;
  const float* ocr    = (const float*)d_in[0];
  const float* obj    = (const float*)d_in[1];
  const float* ocr_dv = (const float*)d_in[2];
  const float* obj_dv = (const float*)d_in[3];
  const float* sa_wq = (const float*)d_in[4];  const float* sa_bq = (const float*)d_in[5];
  const float* sa_wk = (const float*)d_in[6];  const float* sa_bk = (const float*)d_in[7];
  const float* sa_wv = (const float*)d_in[8];  const float* sa_bv = (const float*)d_in[9];
  const float* ln0_g = (const float*)d_in[10]; const float* ln0_b = (const float*)d_in[11];
  const float* qkv_w = (const float*)d_in[12]; const float* qkv_b = (const float*)d_in[13];
  const float* out_w = (const float*)d_in[14]; const float* out_b = (const float*)d_in[15];
  const float* ln1_g = (const float*)d_in[16]; const float* ln1_b = (const float*)d_in[17];
  const float* ff1_w = (const float*)d_in[18]; const float* ff1_b = (const float*)d_in[19];
  const float* ff2_w = (const float*)d_in[20]; const float* ff2_b = (const float*)d_in[21];
  const float* ln2_g = (const float*)d_in[22]; const float* ln2_b = (const float*)d_in[23];

  char* ws = (char*)d_ws;
  float*  xf   = (float*)(ws + OFF_XF);
  __bf16* xb   = (__bf16*)(ws + OFF_XB);
  float*  ldv  = (float*)(ws + OFF_DV);
  __bf16* wb   = (__bf16*)(ws + OFF_WB);
  __bf16* qkvb = (__bf16*)(ws + OFF_QKV);
  __bf16* aob  = (__bf16*)(ws + OFF_AO);
  float*  yb   = (float*)(ws + OFF_Y);
  float*  sb   = (float*)(ws + OFF_S);
  __bf16* pb   = (__bf16*)(ws + OFF_P);

  // setup: concat + bf16 copy, log depths, weight conversion
  concat_x<<<(int)((kTD + 255) / 256), 256, 0, stream>>>(ocr, obj, xf, xb);
  dvlog_k<<<(int)((kT + 255) / 256), 256, 0, stream>>>(ocr_dv, obj_dv, ldv);
  auto cvt = [&](const float* s, long long off, long long n) {
    cvt_f32_bf16<<<(int)((n + 255) / 256), 256, 0, stream>>>(s, wb + off, n);
  };
  cvt(sa_wq, WB_SAWQ, 589824);
  cvt(sa_wk, WB_SAWK, 589824);
  cvt(sa_wv, WB_SAWV, 589824);
  cvt(qkv_w, WB_QKV, 4LL * 2304 * 768);
  cvt(out_w, WB_OUT, 4LL * 768 * 768);
  cvt(ff1_w, WB_FF1, 4LL * 2048 * 768);
  cvt(ff2_w, WB_FF2, 4LL * 768 * 2048);

  const float invSqrtD  = 1.0f / sqrtf((float)kD);
  const float invSqrtDH = 1.0f / sqrtf((float)kDH);

  // ---- semantic attention ----
  gemm(stream, false, true, false, xb, kD, 0, 0, wb + WB_SAWQ, kD, 0, 0,
       qkvb, kD, 0, 0, sa_bq, (int)kT, kD, kD, 1.0f, 1, 1);
  gemm(stream, false, true, false, xb, kD, 0, 0, wb + WB_SAWK, kD, 0, 0,
       qkvb + kTD, kD, 0, 0, sa_bk, (int)kT, kD, kD, 1.0f, 1, 1);
  gemm(stream, false, true, false, xb, kD, 0, 0, wb + WB_SAWV, kD, 0, 0,
       qkvb + 2 * kTD, kD, 0, 0, sa_bv, (int)kT, kD, kD, 1.0f, 1, 1);
  // scores[b] = q[b] k[b]^T / sqrt(D)
  gemm(stream, false, false, false, qkvb, kD, (long long)kN * kD, 0,
       qkvb + kTD, kD, (long long)kN * kD, 0,
       sb, kN, (long long)kN * kN, 0, nullptr, kN, kN, kD, invSqrtD, 1, kB);
  softmax_k<true><<<(int)(kT / 8), 256, 0, stream>>>(sb, pb, ldv);
  // y[b] = P[b] v[b]
  gemm(stream, true, false, false, pb, kN, (long long)kN * kN, 0,
       qkvb + 2 * kTD, kD, (long long)kN * kD, 0,
       yb, kD, (long long)kN * kD, 0, nullptr, kN, kD, kN, 1.0f, 1, kB);
  ln_resid<<<(int)kT, 256, 0, stream>>>(xf, yb, ln0_g, ln0_b, xf, xb);

  // ---- transformer layers ----
  for (int l = 0; l < kL; ++l) {
    gemm(stream, false, true, false, xb, kD, 0, 0,
         wb + WB_QKV + (long long)l * 2304 * 768, kD, 0, 0,
         qkvb, 3 * kD, 0, 0, qkv_b + l * 3 * kD, (int)kT, 3 * kD, kD, 1.0f, 1, 1);
    // scores[b,h] = q_h k_h^T / sqrt(DH)
    gemm(stream, false, false, false, qkvb, 3 * kD, (long long)kN * 3 * kD, kDH,
         qkvb + kD, 3 * kD, (long long)kN * 3 * kD, kDH,
         sb, kN, (long long)kH * kN * kN, (long long)kN * kN,
         nullptr, kN, kN, kDH, invSqrtDH, kH, kB * kH);
    softmax_k<false><<<(int)(kB * kH * kN / 8), 256, 0, stream>>>(sb, pb, nullptr);
    // o_h = P_h v_h
    gemm(stream, true, true, false, pb, kN, (long long)kH * kN * kN, (long long)kN * kN,
         qkvb + 2 * kD, 3 * kD, (long long)kN * 3 * kD, kDH,
         aob, kD, (long long)kN * kD, kDH, nullptr, kN, kDH, kN, 1.0f, kH, kB * kH);
    gemm(stream, false, false, false, aob, kD, 0, 0,
         wb + WB_OUT + (long long)l * 768 * 768, kD, 0, 0,
         yb, kD, 0, 0, out_b + l * kD, (int)kT, kD, kD, 1.0f, 1, 1);
    ln_resid<<<(int)kT, 256, 0, stream>>>(xf, yb, ln1_g + l * kD, ln1_b + l * kD, xf, xb);
    // FFN: ff1 + exact-erf GELU fused into GEMM epilogue, bf16 out -> pb
    gemm(stream, false, true, true, xb, kD, 0, 0,
         wb + WB_FF1 + (long long)l * 2048 * 768, kD, 0, 0,
         pb, kDFF, 0, 0, ff1_b + l * kDFF, (int)kT, kDFF, kD, 1.0f, 1, 1);
    gemm(stream, false, false, false, pb, kDFF, 0, 0,
         wb + WB_FF2 + (long long)l * 768 * 2048, kDFF, 0, 0,
         yb, kD, 0, 0, ff2_b + l * kD, (int)kT, kD, kDFF, 1.0f, 1, 1);
    ln_resid<<<(int)kT, 256, 0, stream>>>(xf, yb, ln2_g + l * kD, ln2_b + l * kD, xf, xb);
  }

  slice_k<<<(int)(((long long)kB * 64 * kD + 255) / 256), 256, 0, stream>>>(xf, (float*)d_out);
}
